// MAGE_Block_47021301956878
// MI455X (gfx1250) — matrix operand order repaired and checked by reading, not compiled, not run on hardware
//
#include <hip/hip_runtime.h>
#include <hip/hip_bf16.h>

// Problem constants
#define NB   32     // batch
#define NN   5000   // nodes
#define NF   64     // feature dim
#define NEK  16     // 2 * expert_num combined (e,k)
#define NG   32     // graph rank d
#define JP   5088   // padded j (159 * 32): 158 processed tiles + 1 prefetch-only tile
#define NKT2 79     // K tile PAIRS processed by k_ugemm (158 tiles)
#define NMP  157    // ceil(313 M-tiles / 2) pairs for k_main

typedef __attribute__((ext_vector_type(16))) __bf16 v16bf;
typedef __attribute__((ext_vector_type(8)))  float  v8f;

union Frag { v16bf v; uint4 q[2]; unsigned u[8]; };

__device__ __forceinline__ v8f zero8() {
  v8f z = {0.f,0.f,0.f,0.f,0.f,0.f,0.f,0.f};
  return z;
}

// float -> bf16 (RNE), duplicated into both halves of a dword
__device__ __forceinline__ unsigned bf16_dup(float f) {
  unsigned x = __float_as_uint(f);
  unsigned r = (x + 0x7FFFu + ((x >> 16) & 1u)) >> 16;
  return r | (r << 16);
}

__device__ __forceinline__ unsigned pk_mul_bf16(unsigned a, unsigned s) {
  unsigned d;
  asm("v_pk_mul_bf16 %0, %1, %2" : "=v"(d) : "v"(a), "v"(s));
  return d;
}

// ---------------- K0: lambda coefficient pair ----------------
__global__ void k_coef(const float* q1, const float* k1, const float* q2,
                       const float* k2, const float* c0, float* cws) {
  __shared__ float r1[64], r2[64];
  int t = threadIdx.x;
  r1[t] = q1[t] * k1[t];
  r2[t] = q2[t] * k2[t];
  __syncthreads();
  for (int s = 32; s; s >>= 1) {
    if (t < s) { r1[t] += r1[t + s]; r2[t] += r2[t + s]; }
    __syncthreads();
  }
  if (t == 0) {
    float lambda_init = 0.8f - 0.6f * __expf(-0.6f);   // depth = 2
    cws[0] = c0[0];
    cws[1] = -(__expf(r1[0]) - __expf(r2[0]) + lambda_init);
  }
}

// ---------------- K1: E2 softmax over j (rows of 5000) -> bf16, padded ----------------
__global__ void k_e2_softmax(const float* __restrict__ E2, __hip_bfloat16* __restrict__ E2s) {
  int r = blockIdx.x;                       // 512 rows = [ek][d]
  const float* in = E2 + (size_t)r * NN;
  __hip_bfloat16* out = E2s + (size_t)r * JP;
  __shared__ float red[256];
  float m = -INFINITY;
  for (int j = threadIdx.x; j < NN; j += 256) m = fmaxf(m, in[j]);
  red[threadIdx.x] = m; __syncthreads();
  for (int s = 128; s; s >>= 1) { if (threadIdx.x < s) red[threadIdx.x] = fmaxf(red[threadIdx.x], red[threadIdx.x + s]); __syncthreads(); }
  m = red[0]; __syncthreads();
  float sum = 0.f;
  for (int j = threadIdx.x; j < NN; j += 256) sum += __expf(in[j] - m);
  red[threadIdx.x] = sum; __syncthreads();
  for (int s = 128; s; s >>= 1) { if (threadIdx.x < s) red[threadIdx.x] += red[threadIdx.x + s]; __syncthreads(); }
  float inv = 1.0f / red[0];
  for (int j = threadIdx.x; j < JP; j += 256)
    out[j] = (j < NN) ? __float2bfloat16(__expf(in[j] - m) * inv) : __float2bfloat16(0.0f);
}

// ---------------- K2: E1 softmax over d (rows of 32) -> bf16 ----------------
__global__ void k_e1_softmax(const float* __restrict__ E1, __hip_bfloat16* __restrict__ E1s) {
  int row  = blockIdx.x * 8 + (threadIdx.x >> 5);    // 80000 rows = [ek][i]
  int lane = threadIdx.x & 31;
  if (row >= NEK * NN) return;
  float v = E1[(size_t)row * NG + lane];
  float m = v;
  for (int s = 16; s; s >>= 1) m = fmaxf(m, __shfl_xor(m, s, 32));
  float e = __expf(v - m);
  float sum = e;
  for (int s = 16; s; s >>= 1) sum += __shfl_xor(sum, s, 32);
  E1s[(size_t)row * NG + lane] = __float2bfloat16(e / sum);
}

// ---------------- K3: x [b][j][f] fp32 -> xb [b][f][j] bf16, padded ----------------
__global__ void k_xt(const float* __restrict__ x, __hip_bfloat16* __restrict__ xb) {
  __shared__ float t[32][65];
  int b = blockIdx.y, j0 = blockIdx.x * 32;
  int tx = threadIdx.x & 63, ty = threadIdx.x >> 6;
  for (int jj = ty; jj < 32; jj += 4) {
    int j = j0 + jj;
    t[jj][tx] = (j < NN) ? x[((size_t)b * NN + j) * NF + tx] : 0.0f;
  }
  __syncthreads();
  int jj = threadIdx.x & 31, f0 = threadIdx.x >> 5;
  for (int f = f0; f < NF; f += 8) {
    xb[((size_t)b * NF + f) * JP + (j0 + jj)] = __float2bfloat16(t[jj][f]);
  }
}

// ---------------- K4: router softmax (fp32) ----------------
__global__ void k_route(const float* __restrict__ x, const float* __restrict__ rw,
                        const float* __restrict__ rb, float* __restrict__ route) {
  int idx = blockIdx.x * 256 + threadIdx.x;          // (b,i) flat
  if (idx >= NB * NN) return;
  const float* xr = x + (size_t)idx * NF;
  float acc[8];
  for (int k = 0; k < 8; k++) acc[k] = rb[k];
  for (int f = 0; f < NF; f++) {
    float xv = xr[f];
    for (int k = 0; k < 8; k++) acc[k] += xv * rw[k * NF + f];
  }
  float m = acc[0];
  for (int k = 1; k < 8; k++) m = fmaxf(m, acc[k]);
  float s = 0.f;
  for (int k = 0; k < 8; k++) { acc[k] = __expf(acc[k] - m); s += acc[k]; }
  float inv = 1.0f / s;
  for (int k = 0; k < 8; k++) route[(size_t)idx * 8 + k] = acc[k] * inv;
}

// ---------------- K5a: u[ek][b][d][f] = E2s[ek] (32 x JP) @ xb (JP x 2048) ----------------
// Double-buffered K loop: fragment loads for tile t+1 stay in flight behind WMMAs of tile t.
__global__ void __launch_bounds__(256) k_ugemm(const __hip_bfloat16* __restrict__ E2s,
                                               const __hip_bfloat16* __restrict__ xb,
                                               float* __restrict__ u) {
  int w = blockIdx.x * 8 + (threadIdx.x >> 5);       // 1024 wave tasks
  int lane = threadIdx.x & 31;
  int ng = w & 31;           // N group of 64 columns == batch b
  int mt = (w >> 5) & 1;     // M tile (d 0..15 / 16..31)
  int ek = w >> 6;
  int half = lane >> 4, lm = lane & 15;
  int ka = half ? 8 : 0;     // A per-lane K offsets: ka..ka+7, ka+16..ka+23
  int kb = half ? 8 : 0;     // B per-lane K offsets: kb..kb+7, kb+16..kb+23
  const __hip_bfloat16* Arow = E2s + ((size_t)ek * NG + mt * 16 + lm) * JP + ka;
  const __hip_bfloat16* Br0 = xb + (size_t)(ng * 64 + 0 * 16 + lm) * JP + kb;
  const __hip_bfloat16* Br1 = xb + (size_t)(ng * 64 + 1 * 16 + lm) * JP + kb;
  const __hip_bfloat16* Br2 = xb + (size_t)(ng * 64 + 2 * 16 + lm) * JP + kb;
  const __hip_bfloat16* Br3 = xb + (size_t)(ng * 64 + 3 * 16 + lm) * JP + kb;

  Frag a0, a1, b0[4], b1[4];
  // preload tile 0
  a0.q[0] = *reinterpret_cast<const uint4*>(Arow);
  a0.q[1] = *reinterpret_cast<const uint4*>(Arow + 16);
  b0[0].q[0] = *reinterpret_cast<const uint4*>(Br0);
  b0[0].q[1] = *reinterpret_cast<const uint4*>(Br0 + 16);
  b0[1].q[0] = *reinterpret_cast<const uint4*>(Br1);
  b0[1].q[1] = *reinterpret_cast<const uint4*>(Br1 + 16);
  b0[2].q[0] = *reinterpret_cast<const uint4*>(Br2);
  b0[2].q[1] = *reinterpret_cast<const uint4*>(Br2 + 16);
  b0[3].q[0] = *reinterpret_cast<const uint4*>(Br3);
  b0[3].q[1] = *reinterpret_cast<const uint4*>(Br3 + 16);

  v8f acc[4] = {zero8(), zero8(), zero8(), zero8()};
#pragma clang loop unroll(disable)
  for (int kt2 = 0; kt2 < NKT2; kt2++) {
    int j1 = kt2 * 64 + 32;
    int j2 = kt2 * 64 + 64;   // final iteration prefetches the zero pad tile (in bounds)
    // prefetch odd tile
    a1.q[0] = *reinterpret_cast<const uint4*>(Arow + j1);
    a1.q[1] = *reinterpret_cast<const uint4*>(Arow + j1 + 16);
    b1[0].q[0] = *reinterpret_cast<const uint4*>(Br0 + j1);
    b1[0].q[1] = *reinterpret_cast<const uint4*>(Br0 + j1 + 16);
    b1[1].q[0] = *reinterpret_cast<const uint4*>(Br1 + j1);
    b1[1].q[1] = *reinterpret_cast<const uint4*>(Br1 + j1 + 16);
    b1[2].q[0] = *reinterpret_cast<const uint4*>(Br2 + j1);
    b1[2].q[1] = *reinterpret_cast<const uint4*>(Br2 + j1 + 16);
    b1[3].q[0] = *reinterpret_cast<const uint4*>(Br3 + j1);
    b1[3].q[1] = *reinterpret_cast<const uint4*>(Br3 + j1 + 16);
    // consume even tile
    acc[0] = __builtin_amdgcn_wmma_f32_16x16x32_bf16(false, a0.v, false, b0[0].v, (short)0, acc[0], false, false);
    acc[1] = __builtin_amdgcn_wmma_f32_16x16x32_bf16(false, a0.v, false, b0[1].v, (short)0, acc[1], false, false);
    acc[2] = __builtin_amdgcn_wmma_f32_16x16x32_bf16(false, a0.v, false, b0[2].v, (short)0, acc[2], false, false);
    acc[3] = __builtin_amdgcn_wmma_f32_16x16x32_bf16(false, a0.v, false, b0[3].v, (short)0, acc[3], false, false);
    // prefetch next even tile
    a0.q[0] = *reinterpret_cast<const uint4*>(Arow + j2);
    a0.q[1] = *reinterpret_cast<const uint4*>(Arow + j2 + 16);
    b0[0].q[0] = *reinterpret_cast<const uint4*>(Br0 + j2);
    b0[0].q[1] = *reinterpret_cast<const uint4*>(Br0 + j2 + 16);
    b0[1].q[0] = *reinterpret_cast<const uint4*>(Br1 + j2);
    b0[1].q[1] = *reinterpret_cast<const uint4*>(Br1 + j2 + 16);
    b0[2].q[0] = *reinterpret_cast<const uint4*>(Br2 + j2);
    b0[2].q[1] = *reinterpret_cast<const uint4*>(Br2 + j2 + 16);
    b0[3].q[0] = *reinterpret_cast<const uint4*>(Br3 + j2);
    b0[3].q[1] = *reinterpret_cast<const uint4*>(Br3 + j2 + 16);
    // consume odd tile
    acc[0] = __builtin_amdgcn_wmma_f32_16x16x32_bf16(false, a1.v, false, b1[0].v, (short)0, acc[0], false, false);
    acc[1] = __builtin_amdgcn_wmma_f32_16x16x32_bf16(false, a1.v, false, b1[1].v, (short)0, acc[1], false, false);
    acc[2] = __builtin_amdgcn_wmma_f32_16x16x32_bf16(false, a1.v, false, b1[2].v, (short)0, acc[2], false, false);
    acc[3] = __builtin_amdgcn_wmma_f32_16x16x32_bf16(false, a1.v, false, b1[3].v, (short)0, acc[3], false, false);
  }
  // store: d = mt*16 + r + half*8, n = ng*64 + nt*16 + lm -> b = ng, f = nt*16+lm
  int b = ng;
#pragma unroll
  for (int nt = 0; nt < 4; nt++) {
    int f = nt * 16 + lm;
#pragma unroll
    for (int r = 0; r < 8; r++) {
      int d = mt * 16 + r + half * 8;
      u[(((size_t)ek * NB + b) * NG + d) * NF + f] = acc[nt][r];
    }
  }
}

// ---------------- K5b: fold outer_w and coef[e]: u2[ek][b][f][d] (bf16) ----------------
__global__ void k_fold(const float* __restrict__ u, const float* __restrict__ ow,
                       const float* __restrict__ cws, __hip_bfloat16* __restrict__ u2) {
  int idx = blockIdx.x * 256 + threadIdx.x;          // [ek][b][f][d]
  if (idx >= NEK * NB * NF * NG) return;
  int d = idx & 31, f = (idx >> 5) & 63, b = (idx >> 11) & 31, ek = idx >> 16;
  const float* ur = u + (((size_t)ek * NB + b) * NG + d) * NF;
  const float* wr = ow + (size_t)f * NF;
  float s = 0.f;
  for (int ff = 0; ff < NF; ff++) s += ur[ff] * wr[ff];
  u2[idx] = __float2bfloat16(s * cws[ek >> 3]);
}

// ---------------- K6: out[b][i][f] = sum_ek (route * E1s) @ u2 + outer_b ----------------
// 2 M-tiles x 4 N-tiles per wave: B fragments reused across both M-tiles.
__global__ void __launch_bounds__(256) k_main(const __hip_bfloat16* __restrict__ E1s,
                                              const __hip_bfloat16* __restrict__ u2,
                                              const float* __restrict__ route,
                                              const float* __restrict__ ob,
                                              float* __restrict__ out) {
  int mtp = blockIdx.x * 8 + (threadIdx.x >> 5);     // M-tile pair index
  if (mtp >= NMP) return;
  int b = blockIdx.y;
  int lane = threadIdx.x & 31;
  int half = lane >> 4, lm = lane & 15;
  int ka = half ? 8 : 0;
  int kb = half ? 8 : 0;
  int mt0 = mtp * 2, mt1 = mt0 + 1;
  int i0 = mt0 * 16 + lm;
  int i1 = mt1 * 16 + lm;
  bool ok0 = i0 < NN, ok1 = i1 < NN;
  int i0c = ok0 ? i0 : (NN - 1);           // clamp for safe loads; scale forced to 0
  int i1c = ok1 ? i1 : (NN - 1);
  const float* rrow0 = route + ((size_t)b * NN + i0c) * 8;
  const float* rrow1 = route + ((size_t)b * NN + i1c) * 8;

  v8f acc[8] = {zero8(), zero8(), zero8(), zero8(),
                zero8(), zero8(), zero8(), zero8()};
#pragma clang loop unroll(disable)
  for (int k = 0; k < 8; k++) {
    unsigned s0 = bf16_dup(ok0 ? rrow0[k] : 0.0f);   // coef[e] already folded in u2
    unsigned s1 = bf16_dup(ok1 ? rrow1[k] : 0.0f);
#pragma unroll
    for (int e = 0; e < 2; e++) {
      int ek = e * 8 + k;
      const __hip_bfloat16* A0 = E1s + ((size_t)ek * NN + i0c) * NG;
      const __hip_bfloat16* A1 = E1s + ((size_t)ek * NN + i1c) * NG;
      Frag a0, a1;
      a0.q[0] = *reinterpret_cast<const uint4*>(A0 + ka);
      a0.q[1] = *reinterpret_cast<const uint4*>(A0 + ka + 16);
      a1.q[0] = *reinterpret_cast<const uint4*>(A1 + ka);
      a1.q[1] = *reinterpret_cast<const uint4*>(A1 + ka + 16);
#pragma unroll
      for (int c = 0; c < 8; c++) {
        a0.u[c] = pk_mul_bf16(a0.u[c], s0);
        a1.u[c] = pk_mul_bf16(a1.u[c], s1);
      }
#pragma unroll
      for (int nt = 0; nt < 4; nt++) {
        int f = nt * 16 + lm;
        const __hip_bfloat16* Brow = u2 + (((size_t)ek * NB + b) * NF + f) * NG;
        Frag bb;
        bb.q[0] = *reinterpret_cast<const uint4*>(Brow + kb);
        bb.q[1] = *reinterpret_cast<const uint4*>(Brow + kb + 16);
        acc[nt] = __builtin_amdgcn_wmma_f32_16x16x32_bf16(
            false, a0.v, false, bb.v, (short)0, acc[nt], false, false);
        acc[4 + nt] = __builtin_amdgcn_wmma_f32_16x16x32_bf16(
            false, a1.v, false, bb.v, (short)0, acc[4 + nt], false, false);
      }
    }
  }
#pragma unroll
  for (int nt = 0; nt < 4; nt++) {
    int f = nt * 16 + lm;
    float bias = ob[f];
#pragma unroll
    for (int r = 0; r < 8; r++) {
      int ii = mt0 * 16 + r + half * 8;
      if (ii < NN) out[((size_t)b * NN + ii) * NF + f] = acc[nt][r] + bias;
    }
#pragma unroll
    for (int r = 0; r < 8; r++) {
      int ii = mt1 * 16 + r + half * 8;
      if (ii < NN) out[((size_t)b * NN + ii) * NF + f] = acc[4 + nt][r] + bias;
    }
  }
}

extern "C" void kernel_launch(void* const* d_in, const int* in_sizes, int n_in,
                              void* d_out, int out_size, void* d_ws, size_t ws_size,
                              hipStream_t stream) {
  const float* x   = (const float*)d_in[0];
  const float* E1  = (const float*)d_in[1];
  const float* E2  = (const float*)d_in[2];
  const float* rw  = (const float*)d_in[3];
  const float* rb  = (const float*)d_in[4];
  const float* ow  = (const float*)d_in[5];
  const float* obv = (const float*)d_in[6];
  const float* lq1 = (const float*)d_in[7];
  const float* lk1 = (const float*)d_in[8];
  const float* lq2 = (const float*)d_in[9];
  const float* lk2 = (const float*)d_in[10];
  const float* c0  = (const float*)d_in[11];
  float* out = (float*)d_out;

  // workspace carve (256B aligned)
  char* ws = (char*)d_ws;
  size_t off = 0;
  auto carve = [&](size_t bytes) { char* p = ws + off; off = (off + bytes + 255) & ~(size_t)255; return p; };
  float*          cws   = (float*)carve(2 * sizeof(float));
  float*          route = (float*)carve((size_t)NB * NN * 8 * sizeof(float));
  __hip_bfloat16* E2s   = (__hip_bfloat16*)carve((size_t)NEK * NG * JP * sizeof(__hip_bfloat16));
  __hip_bfloat16* E1s   = (__hip_bfloat16*)carve((size_t)NEK * NN * NG * sizeof(__hip_bfloat16));
  __hip_bfloat16* xb    = (__hip_bfloat16*)carve((size_t)NB * NF * JP * sizeof(__hip_bfloat16));
  float*          u     = (float*)carve((size_t)NEK * NB * NG * NF * sizeof(float));
  __hip_bfloat16* u2    = (__hip_bfloat16*)carve((size_t)NEK * NB * NF * NG * sizeof(__hip_bfloat16));

  k_coef<<<1, 64, 0, stream>>>(lq1, lk1, lq2, lk2, c0, cws);
  k_e2_softmax<<<NEK * NG, 256, 0, stream>>>(E2, E2s);
  k_e1_softmax<<<(NEK * NN) / 8, 256, 0, stream>>>(E1, E1s);
  k_xt<<<dim3(JP / 32, NB), 256, 0, stream>>>(x, xb);
  k_route<<<(NB * NN + 255) / 256, 256, 0, stream>>>(x, rw, rb, route);
  k_ugemm<<<(NEK * 2 * 32) / 8, 256, 0, stream>>>(E2s, xb, u);
  k_fold<<<(NEK * NB * NF * NG) / 256, 256, 0, stream>>>(u, ow, cws, u2);
  k_main<<<dim3((NMP + 7) / 8, NB), 256, 0, stream>>>(E1s, u2, route, obv, out);
}
